// MOEMlp_10797547782275
// MI455X (gfx1250) — matrix operand order repaired and checked by reading, not compiled, not run on hardware
//
#include <hip/hip_runtime.h>
#include <hip/hip_bf16.h>
#include <math.h>

// Problem constants (match reference)
#define D_IN   1024
#define H_DIM  4096
#define E_NUM  23
#define K_TOP  3
#define NTOK   8192          // 4 * 2048 tokens
#define TILE_M 16            // tokens per workgroup tile
#define HCHUNK 128           // hidden columns per fused chunk

// Packed fragment geometry: one 32x16 (KxN) bf16 B-tile = 32 lanes x 32 bytes = 1KB.
// Per expert: W1 frag buf = (H/16 ntiles)*(D/32 ktiles)*1KB = 8 MB, W2 same.
#define FRAG_BYTES   1024
#define W_FRAG_EXP   (8u * 1024u * 1024u)     // per-expert packed bytes (both W1 & W2)

typedef __attribute__((ext_vector_type(16))) __bf16        v16bf;
typedef __attribute__((ext_vector_type(8)))  float         v8f;
typedef __attribute__((ext_vector_type(4)))  float         v4f;
typedef __attribute__((ext_vector_type(4)))  unsigned int  v4u;

union BF16x16 { unsigned int u[8]; v4u q[2]; v16bf v; };

__device__ __forceinline__ unsigned short f2bf(float f) {
  unsigned int u = __builtin_bit_cast(unsigned int, f);
  u += 0x7fffu + ((u >> 16) & 1u);          // round-to-nearest-even
  return (unsigned short)(u >> 16);
}
__device__ __forceinline__ unsigned int pack_bf2(float a, float b) {
  return (unsigned int)f2bf(a) | ((unsigned int)f2bf(b) << 16);
}
__device__ __forceinline__ float gelu_exact(float h) {
  return 0.5f * h * (1.0f + erff(h * 0.70710678118654752f));
}

// ---------------------------------------------------------------------------
// Kernel 1: zero output and routing counters
// ---------------------------------------------------------------------------
__global__ void init_kernel(float* __restrict__ out, int total,
                            int* __restrict__ counts, int* __restrict__ cursors,
                            float* __restrict__ Psum) {
  int i = blockIdx.x * blockDim.x + threadIdx.x;
  for (int j = i; j < total; j += gridDim.x * blockDim.x) out[j] = 0.0f;
  if (blockIdx.x == 0 && threadIdx.x < 32) {
    if (threadIdx.x < E_NUM + 1) { counts[threadIdx.x] = 0; cursors[threadIdx.x] = 0; }
    if (threadIdx.x < E_NUM)     { Psum[threadIdx.x] = 0.0f; }
  }
}

// ---------------------------------------------------------------------------
// Kernel 2: pack one [K,N] f32 matrix per expert into bf16 WMMA B-fragment
// order.  Fragment (ntile, ktile): 32 lanes x 16 bf16; lane holds
// n = ntile*16 + (lane&15), k = ktile*32 + (lane>>4)*8 + {0..7, 16..23 as K pairs}.
// grid = (slots/256, 24); blockIdx.y == 23 -> shared-expert matrix.
// ---------------------------------------------------------------------------
__global__ __launch_bounds__(256) void pack_weights_kernel(
    const float* __restrict__ W, const float* __restrict__ sW,
    char* __restrict__ dst, int K, int N, size_t eStride) {
  const int e = blockIdx.y;
  const float* src = (e == E_NUM) ? sW : W + (size_t)e * eStride;
  const int slot = blockIdx.x * 256 + threadIdx.x;   // (ntile*Kt + ktile)*32 + lane
  const int Kt = K >> 5;
  const int lane = slot & 31;
  const int kt   = (slot >> 5) % Kt;
  const int nt   = (slot >> 5) / Kt;
  const int n    = nt * 16 + (lane & 15);
  const int kb   = kt * 32 + (lane >> 4) * 8;
  unsigned int u[8];
  #pragma unroll
  for (int v = 0; v < 8; ++v) {
    int k0 = kb + 2 * v + (v >= 4 ? 8 : 0);
    u[v] = pack_bf2(src[(size_t)k0 * N + n], src[(size_t)(k0 + 1) * N + n]);
  }
  v4u* dp = (v4u*)(dst + (size_t)e * W_FRAG_EXP + (size_t)slot * 32);
  v4u q0 = { u[0], u[1], u[2], u[3] };
  v4u q1 = { u[4], u[5], u[6], u[7] };
  dp[0] = q0; dp[1] = q1;
}

// ---------------------------------------------------------------------------
// Kernel 3: gate = sigmoid(x @ gate_W + gate_b); top-3 on (gate + bias);
// one wave32 per token.
// ---------------------------------------------------------------------------
__global__ __launch_bounds__(256) void gate_kernel(
    const float* __restrict__ x, const float* __restrict__ gW,
    const float* __restrict__ gb, const float* __restrict__ bias,
    int* __restrict__ topk_idx, float* __restrict__ topk_w,
    int* __restrict__ counts, float* __restrict__ Psum) {
  const int tok  = (blockIdx.x * 256 + threadIdx.x) >> 5;
  const int lane = threadIdx.x & 31;
  if (tok >= NTOK) return;
  const float* xr = x + (size_t)tok * D_IN;

  float g[E_NUM];
  #pragma unroll
  for (int e = 0; e < E_NUM; ++e) g[e] = 0.0f;

  for (int j = 0; j < D_IN / 32; ++j) {
    int d = j * 32 + lane;
    float xv = xr[d];
    const float* gr = gW + (size_t)d * E_NUM;
    #pragma unroll
    for (int e = 0; e < E_NUM; ++e) g[e] += xv * gr[e];
  }
  #pragma unroll
  for (int e = 0; e < E_NUM; ++e) {
    float v = g[e];
    for (int off = 16; off > 0; off >>= 1) v += __shfl_xor(v, off, 32);
    g[e] = 1.0f / (1.0f + expf(-(v + gb[e])));
  }
  if (lane == 0) {
    float sumg = 0.0f;
    #pragma unroll
    for (int e = 0; e < E_NUM; ++e) sumg += g[e];
    float inv = 1.0f / sumg;
    #pragma unroll
    for (int e = 0; e < E_NUM; ++e) atomicAdd(&Psum[e], g[e] * inv);

    float key[E_NUM];
    #pragma unroll
    for (int e = 0; e < E_NUM; ++e) key[e] = g[e] + bias[e];
    int   idx[K_TOP];
    float wv[K_TOP];
    float tsum = 0.0f;
    #pragma unroll
    for (int k = 0; k < K_TOP; ++k) {
      int bi = 0; float bv = -3.0e38f;
      #pragma unroll
      for (int e = 0; e < E_NUM; ++e)
        if (key[e] > bv) { bv = key[e]; bi = e; }
      idx[k] = bi; wv[k] = g[bi]; key[bi] = -3.0e38f; tsum += g[bi];
    }
    float itw = 1.0f / tsum;
    #pragma unroll
    for (int k = 0; k < K_TOP; ++k) {
      topk_idx[tok * K_TOP + k] = idx[k];
      topk_w[tok * K_TOP + k]   = wv[k] * itw;
      atomicAdd(&counts[idx[k]], 1);
    }
  }
}

// ---------------------------------------------------------------------------
// Kernel 4: segment offsets + aux loss scalar
// ---------------------------------------------------------------------------
__global__ void finalize_gate_kernel(const int* __restrict__ counts,
                                     const float* __restrict__ Psum,
                                     int* __restrict__ offsets,
                                     float* __restrict__ aux_out) {
  if (threadIdx.x == 0 && blockIdx.x == 0) {
    int off = 0;
    for (int e = 0; e < E_NUM; ++e) { offsets[e] = off; off += counts[e]; }
    offsets[E_NUM] = off;
    float aux = 0.0f;
    for (int e = 0; e < E_NUM; ++e) {
      float P = Psum[e] / (float)NTOK;
      float F = (float)E_NUM * (float)counts[e] / ((float)K_TOP * (float)NTOK);
      aux += P * F;
    }
    *aux_out = aux;
  }
}

// ---------------------------------------------------------------------------
// Kernel 5: scatter tokens into per-expert segments
// ---------------------------------------------------------------------------
__global__ void scatter_kernel(const int* __restrict__ topk_idx,
                               const float* __restrict__ topk_w,
                               const int* __restrict__ offsets,
                               int* __restrict__ cursors,
                               int* __restrict__ seg_tok,
                               float* __restrict__ seg_w) {
  int i = blockIdx.x * blockDim.x + threadIdx.x;
  if (i >= NTOK * K_TOP) return;
  int e   = topk_idx[i];
  int pos = offsets[e] + atomicAdd(&cursors[e], 1);
  seg_tok[pos] = i / K_TOP;
  seg_w[pos]   = topk_w[i];
}

// ---------------------------------------------------------------------------
// Kernel 6: fused grouped-GEMM MoE MLP with bf16 WMMA on packed fragments.
// grid = (max_tiles, 24); blockIdx.y == 23 -> shared expert over all tokens.
// 256 threads = 8 wave32s. Per tile: 16 tokens x full D->H->D MLP.
// Steady state per WMMA: 2x ds_load_b128 (A) + 2x global_load_b128 (B).
// ---------------------------------------------------------------------------
__global__ __launch_bounds__(256) void moe_mlp_kernel(
    const float* __restrict__ x,
    const char* __restrict__ pW1,     // packed bf16 W1 fragments [24][8MB]
    const char* __restrict__ pW2,     // packed bf16 W2 fragments [24][8MB]
    const float* __restrict__ b1, const float* __restrict__ b2,
    const float* __restrict__ sb1, const float* __restrict__ sb2,
    const int* __restrict__ counts, const int* __restrict__ offsets,
    const int* __restrict__ seg_tok, const float* __restrict__ seg_w,
    float* __restrict__ out) {
  const int e = blockIdx.y;
  const bool se = (e == E_NUM);                 // shared expert
  const int cnt  = se ? NTOK : counts[e];
  const int base = se ? 0    : offsets[e];
  if ((int)blockIdx.x * TILE_M >= cnt) return;

  const char* w1f = pW1 + (size_t)e * W_FRAG_EXP;
  const char* w2f = pW2 + (size_t)e * W_FRAG_EXP;
  const float* b1p = se ? sb1 : b1 + (size_t)e * H_DIM;
  const float* b2p = se ? sb2 : b2 + (size_t)e * D_IN;

  // X tile in A-fragment order: 32 ktiles x 32 lanes x 32 bytes = 32 KB
  __shared__ v4u XsF[2048];
  // hidden chunk in A-fragment order: 4 ktiles x 32 lanes x 32 bytes = 4 KB
  __shared__ __align__(16) unsigned short HsF[2048];
  __shared__ int   tokS[TILE_M];
  __shared__ float wS[TILE_M];

  const int tid = threadIdx.x;
  if (tid < TILE_M) {
    int r = blockIdx.x * TILE_M + tid;
    if (r < cnt) {
      tokS[tid] = se ? r : seg_tok[base + r];
      wS[tid]   = se ? 1.0f : seg_w[base + r];
    } else { tokS[tid] = -1; wS[tid] = 0.0f; }
  }
  __syncthreads();

  // ---- Stage X tile into LDS in A-fragment order (slot = kt*32 + lane) -----
  for (int s = tid; s < 1024; s += 256) {
    const int ln = s & 31, kt = s >> 5;
    const int m = ln & 15, kh = ln >> 4;
    const int t = tokS[m];
    unsigned int u[8];
    if (t >= 0) {
      const float* xr = x + (size_t)t * D_IN + kt * 32 + kh * 8;
      v4f f0 = *(const v4f*)(xr);        // k + 0..3
      v4f f1 = *(const v4f*)(xr + 4);    // k + 4..7
      v4f f2 = *(const v4f*)(xr + 16);   // k + 16..19
      v4f f3 = *(const v4f*)(xr + 20);   // k + 20..23
      u[0] = pack_bf2(f0.x, f0.y); u[1] = pack_bf2(f0.z, f0.w);
      u[2] = pack_bf2(f1.x, f1.y); u[3] = pack_bf2(f1.z, f1.w);
      u[4] = pack_bf2(f2.x, f2.y); u[5] = pack_bf2(f2.z, f2.w);
      u[6] = pack_bf2(f3.x, f3.y); u[7] = pack_bf2(f3.z, f3.w);
    } else {
      #pragma unroll
      for (int v = 0; v < 8; ++v) u[v] = 0u;
    }
    v4u q0 = { u[0], u[1], u[2], u[3] };
    v4u q1 = { u[4], u[5], u[6], u[7] };
    XsF[s * 2 + 0] = q0;
    XsF[s * 2 + 1] = q1;
  }
  __syncthreads();

  const int lane = tid & 31;
  const int wid  = tid >> 5;
  const int hi   = lane >> 4;     // half-wave (K split per ISA fragment layout)
  const int lm   = lane & 15;

  v8f acc[8] = {};                // 8 x [16x16] f32 output tiles (128 cols/wave)

  for (int hc = 0; hc < H_DIM; hc += HCHUNK) {
    // ---- Phase 1: h[16, HCHUNK] = gelu(X @ W1 + b1); 16 cols per wave -----
    const int nt1 = (hc >> 4) + wid;              // W1 ntile for this wave
    v8f c1 = {};
    for (int kt = 0; kt < D_IN / 32; ++kt) {
      BF16x16 a, b;
      const v4u* pa = &XsF[(kt * 32 + lane) * 2];
      a.q[0] = pa[0]; a.q[1] = pa[1];
      const v4u* pb = (const v4u*)(w1f + (((size_t)nt1 * 32 + kt) << 10) + lane * 32);
      b.q[0] = pb[0]; b.q[1] = pb[1];
      __builtin_prefetch(w1f + (((size_t)nt1 * 32 + kt + 1) << 10) + lane * 32, 0, 1);
      c1 = __builtin_amdgcn_wmma_f32_16x16x32_bf16(
          false, a.v, false, b.v, (short)0, c1, false, false);
    }
    __syncthreads();                       // prior readers of HsF are done
    {
      const int kcol = wid * 16 + lm;      // chunk-local hidden column
      const int kt   = kcol >> 5;
      const int r5   = kcol & 31;
      const int grp  = r5 >> 4;            // i>>3
      const int w16  = r5 & 15;
      const int kh2  = w16 >> 3;
      const int i7   = w16 & 7;
      const float bb = b1p[hc + kcol];
      #pragma unroll
      for (int v = 0; v < 8; ++v) {
        int m  = v + 8 * hi;               // C layout: VGPR v -> row, half +8
        float h = c1[v] + bb;
        HsF[kt * 512 + (kh2 * 16 + m) * 16 + grp * 8 + i7] = f2bf(gelu_exact(h));
      }
    }
    __syncthreads();

    // ---- Phase 2: acc += h_chunk @ W2[hc:hc+HCHUNK, :]; 128 cols/wave -----
    BF16x16 af[4];
    #pragma unroll
    for (int t = 0; t < 4; ++t) {
      const v4u* pa = (const v4u*)((const char*)HsF + t * 1024 + lane * 32);
      af[t].q[0] = pa[0]; af[t].q[1] = pa[1];
    }
    const int ktb = hc >> 5;                       // base W2 ktile for chunk
    #pragma unroll
    for (int j = 0; j < 8; ++j) {
      const int nt2 = wid * 8 + j;                 // W2 ntile
      #pragma unroll
      for (int t = 0; t < 4; ++t) {
        BF16x16 b;
        const v4u* pb = (const v4u*)(w2f + (((size_t)nt2 * 128 + ktb + t) << 10) + lane * 32);
        b.q[0] = pb[0]; b.q[1] = pb[1];
        acc[j] = __builtin_amdgcn_wmma_f32_16x16x32_bf16(
            false, af[t].v, false, b.v, (short)0, acc[j], false, false);
      }
    }
  }

  // ---- Epilogue: out[token] += w * (acc + b2) -----------------------------
  #pragma unroll
  for (int j = 0; j < 8; ++j) {
    const int dcol0 = wid * 128 + j * 16;
    #pragma unroll
    for (int v = 0; v < 8; ++v) {
      int m = v + 8 * hi;
      int t = tokS[m];
      if (t >= 0) {
        float val = wS[m] * (acc[j][v] + b2p[dcol0 + lm]);
        atomicAdd(&out[(size_t)t * D_IN + dcol0 + lm], val);
      }
    }
  }
}

// ---------------------------------------------------------------------------
// Host launcher
// ---------------------------------------------------------------------------
extern "C" void kernel_launch(void* const* d_in, const int* in_sizes, int n_in,
                              void* d_out, int out_size, void* d_ws, size_t ws_size,
                              hipStream_t stream) {
  const float* x    = (const float*)d_in[0];
  const float* gW   = (const float*)d_in[1];
  const float* gb   = (const float*)d_in[2];
  const float* W1   = (const float*)d_in[3];
  const float* b1   = (const float*)d_in[4];
  const float* W2   = (const float*)d_in[5];
  const float* b2   = (const float*)d_in[6];
  const float* sW1  = (const float*)d_in[7];
  const float* sb1  = (const float*)d_in[8];
  const float* sW2  = (const float*)d_in[9];
  const float* sb2  = (const float*)d_in[10];
  const float* bias = (const float*)d_in[11];

  float* out = (float*)d_out;                 // [NTOK*D_IN] out, then [1] aux

  char* ws = (char*)d_ws;
  int*   topk_idx = (int*)  (ws + 0);                       // 3N ints
  float* topk_w   = (float*)(ws + 98304);                   // 3N floats
  int*   seg_tok  = (int*)  (ws + 196608);                  // 3N ints
  float* seg_w    = (float*)(ws + 294912);                  // 3N floats
  int*   counts   = (int*)  (ws + 393216);                  // 24 ints
  int*   offsets  = (int*)  (ws + 393216 + 128);            // 24 ints
  int*   cursors  = (int*)  (ws + 393216 + 256);            // 24 ints
  float* Psum     = (float*)(ws + 393216 + 384);            // 23 floats
  char*  pW1      = ws + (1u << 20);                        // 24 * 8 MB packed bf16
  char*  pW2      = pW1 + (size_t)24 * W_FRAG_EXP;          // 24 * 8 MB packed bf16

  init_kernel<<<2048, 256, 0, stream>>>(out, NTOK * D_IN, counts, cursors, Psum);

  {   // pack W1 (+shared) and W2 (+shared) into WMMA B-fragment order
    dim3 pg(1024, E_NUM + 1);   // 262144 slots / 256 per block
    pack_weights_kernel<<<pg, 256, 0, stream>>>(W1, sW1, pW1, D_IN, H_DIM,
                                                (size_t)D_IN * H_DIM);
    pack_weights_kernel<<<pg, 256, 0, stream>>>(W2, sW2, pW2, H_DIM, D_IN,
                                                (size_t)H_DIM * D_IN);
  }

  gate_kernel<<<NTOK / 8, 256, 0, stream>>>(x, gW, gb, bias,
                                            topk_idx, topk_w, counts, Psum);
  finalize_gate_kernel<<<1, 32, 0, stream>>>(counts, Psum, offsets,
                                             out + (size_t)NTOK * D_IN);
  scatter_kernel<<<(NTOK * K_TOP + 255) / 256, 256, 0, stream>>>(
      topk_idx, topk_w, offsets, cursors, seg_tok, seg_w);

  dim3 grid(NTOK / TILE_M, E_NUM + 1);        // 512 tiles x (23 experts + shared)
  moe_mlp_kernel<<<grid, 256, 0, stream>>>(x, pW1, pW2, b1, b2, sb1, sb2,
                                           counts, offsets, seg_tok, seg_w, out);
}
